// STICKYLlamaAttention_14130442404058
// MI455X (gfx1250) — matrix operand order repaired and checked by reading, not compiled, not run on hardware
//
#include <hip/hip_runtime.h>
#include <hip/hip_bf16.h>
#include <math.h>

// ---------------------------------------------------------------------------
// LLaMA attention prefill on gfx1250 (MI455X):
//   f32 -> f16 convert, WMMA NT GEMMs for Q/K/V/O projections,
//   fused flash-attention with online softmax, all matrix math on
//   v_wmma_f32_16x16x32_f16 (wave32).  Global->LDS staging uses the CDNA5
//   async copy path (GLOBAL_LOAD_ASYNC_TO_LDS_B128 + s_wait_asynccnt) when
//   the toolchain exposes it, else batched load/store.
// ---------------------------------------------------------------------------

#define S_LEN   2048
#define HID     4096
#define DHEAD   128
#define NQH     32
#define NKVH    8

typedef __attribute__((ext_vector_type(16))) _Float16 v16h;
typedef __attribute__((ext_vector_type(8)))  float    v8f;

struct __align__(16) h8 { _Float16 h[8]; };
union F16x16 { v16h v; _Float16 h[16]; };

#if __has_builtin(__builtin_amdgcn_global_load_async_to_lds_b128)
#define HAVE_ASYNC_LDS 1
#else
#define HAVE_ASYNC_LDS 0
#endif

#if HAVE_ASYNC_LDS
// Per-lane async 16B copy global -> LDS (ASYNCcnt-tracked, no VGPR bounce).
// Builtin parameter types (from clang diagnostic): v4i in AS(1) / AS(3).
// Flat LDS address low 32 bits == LDS byte offset (aperture rules, ISA 10.2);
// flat address == global address for AS(1).
typedef int v4i __attribute__((vector_size(16)));
__device__ __forceinline__ void async_cp16(void* lds, const void* g) {
  __builtin_amdgcn_global_load_async_to_lds_b128(
      (__attribute__((address_space(1))) v4i*)(unsigned long long)g,
      (__attribute__((address_space(3))) v4i*)(unsigned int)(unsigned long long)lds,
      0, 0);
}
__device__ __forceinline__ void wait_async0() {
#if __has_builtin(__builtin_amdgcn_s_wait_asynccnt)
  __builtin_amdgcn_s_wait_asynccnt(0);
#else
  asm volatile("s_wait_asynccnt 0x0" ::: "memory");
#endif
}
#endif

// A-fragment (16x32 f16, MxK): lane l holds row m = l&15; elements are
// 16 consecutive K values starting at 2*kB (0 for lanes 0-15, 16 for 16-31).
__device__ __forceinline__ v16h load_frag_a(const _Float16* row, int kB) {
  h8 lo = *(const h8*)(row + 2 * kB);
  h8 hi = *(const h8*)(row + 2 * kB + 8);
  F16x16 u;
#pragma unroll
  for (int i = 0; i < 8; ++i) { u.h[i] = lo.h[i]; u.h[8 + i] = hi.h[i]; }
  return u.v;
}

// B-fragment (32x16 f16, KxN): lane l holds column n = l&15; elements are
// 16 consecutive K values starting at kB2 (0 for lanes 0-15, 16 for 16-31).
__device__ __forceinline__ v16h load_frag_b(const _Float16* p) {
  h8 lo = *(const h8*)p;
  h8 hi = *(const h8*)(p + 8);
  F16x16 u;
#pragma unroll
  for (int i = 0; i < 8; ++i) { u.h[i] = lo.h[i]; u.h[8 + i] = hi.h[i]; }
  return u.v;
}

__device__ __forceinline__ v8f wmma_f16(v16h a, v16h b, v8f c) {
  return __builtin_amdgcn_wmma_f32_16x16x32_f16(false, a, false, b,
                                                (short)0, c, false, false);
}

// ---------------------------------------------------------------------------
// Elementwise f32 -> f16
// ---------------------------------------------------------------------------
__global__ __launch_bounds__(256) void cvt_f32_to_f16(const float* __restrict__ src,
                                                      _Float16* __restrict__ dst,
                                                      int n) {
  int i = blockIdx.x * 256 + threadIdx.x;
  if (i < n) dst[i] = (_Float16)src[i];
}

// ---------------------------------------------------------------------------
// NT GEMM: C[m,n] = sum_k A[m,k] * B[n,k]   (A: MxK f16, B: NxK f16, C f32)
// Block tile 128(M) x 128(N), 256 threads = 8 waves (4 M-tiles x 2 N-halves),
// each wave computes a 32x64 slab: per 32-deep K-step 2 A-frags + 4 B-frags
// feed 8 WMMAs.  Staging uses async global->LDS copies when available.
// ---------------------------------------------------------------------------
__global__ __launch_bounds__(256) void gemm_nt(const _Float16* __restrict__ A,
                                               const _Float16* __restrict__ B,
                                               float* __restrict__ C,
                                               int M, int N, int K) {
  __shared__ __align__(16) _Float16 At[128][40];   // 32 cols used, padded row
  __shared__ __align__(16) _Float16 Bt[128][40];

  const int tid  = threadIdx.x;
  const int lane = tid & 31;
  const int wave = tid >> 5;
  const int wm   = wave & 3;          // M sub-tile 0..3 (32 rows each)
  const int wn   = wave >> 2;         // N half 0..1 (64 cols each)
  const int mBase = blockIdx.y * 128;
  const int nBase = blockIdx.x * 128;

  const int ln15 = lane & 15;
  const int hi   = lane >> 4;
  const int kBa  = hi ? 8 : 0;        // A-fragment K base
  const int kBb  = hi ? 16 : 0;       // B-fragment K base

  v8f acc[2][4];
#pragma unroll
  for (int s = 0; s < 2; ++s)
#pragma unroll
    for (int t = 0; t < 4; ++t) { v8f z = {}; acc[s][t] = z; }

  // staging coordinates: 128x32 tile = 512 b128 chunks, 2 per thread per tile
  const int r0 = tid >> 2;
  const int c80 = (tid & 3) * 8;
  const int r1 = (tid + 256) >> 2;
  const int c81 = ((tid + 256) & 3) * 8;

  for (int k0 = 0; k0 < K; k0 += 32) {
    __syncthreads();
#if HAVE_ASYNC_LDS
    async_cp16(&At[r0][c80], A + (size_t)(mBase + r0) * K + k0 + c80);
    async_cp16(&At[r1][c81], A + (size_t)(mBase + r1) * K + k0 + c81);
    async_cp16(&Bt[r0][c80], B + (size_t)(nBase + r0) * K + k0 + c80);
    async_cp16(&Bt[r1][c81], B + (size_t)(nBase + r1) * K + k0 + c81);
    wait_async0();
#else
    // batch all global loads, then all LDS stores (no per-chunk stall)
    h8 ta0 = *(const h8*)(A + (size_t)(mBase + r0) * K + k0 + c80);
    h8 ta1 = *(const h8*)(A + (size_t)(mBase + r1) * K + k0 + c81);
    h8 tb0 = *(const h8*)(B + (size_t)(nBase + r0) * K + k0 + c80);
    h8 tb1 = *(const h8*)(B + (size_t)(nBase + r1) * K + k0 + c81);
    if (k0 + 32 < K) {               // prefetch next K-tile into caches
      __builtin_prefetch(A + (size_t)(mBase + r0) * K + k0 + 32 + c80, 0, 1);
      __builtin_prefetch(B + (size_t)(nBase + r0) * K + k0 + 32 + c80, 0, 1);
    }
    *(h8*)&At[r0][c80] = ta0;
    *(h8*)&At[r1][c81] = ta1;
    *(h8*)&Bt[r0][c80] = tb0;
    *(h8*)&Bt[r1][c81] = tb1;
#endif
    __syncthreads();

    v16h aF[2];
#pragma unroll
    for (int s = 0; s < 2; ++s)
      aF[s] = load_frag_a(&At[wm * 32 + s * 16 + ln15][0], kBa);
#pragma unroll
    for (int t = 0; t < 4; ++t) {
      v16h bF = load_frag_b(&Bt[wn * 64 + t * 16 + ln15][kBb]);
#pragma unroll
      for (int s = 0; s < 2; ++s)
        acc[s][t] = wmma_f16(aF[s], bF, acc[s][t]);
    }
  }

  // epilogue: C/D layout — lane holds column n, VGPR r holds row r + 8*hi
  const int n0 = nBase + wn * 64;
#pragma unroll
  for (int s = 0; s < 2; ++s) {
#pragma unroll
    for (int t = 0; t < 4; ++t) {
#pragma unroll
      for (int r = 0; r < 8; ++r) {
        int m = mBase + wm * 32 + s * 16 + r + hi * 8;
        int n = n0 + t * 16 + ln15;
        C[(size_t)m * N + n] = acc[s][t][r];
      }
    }
  }
}

// ---------------------------------------------------------------------------
// RoPE + pack kernels
// ---------------------------------------------------------------------------
#define LN10000_OVER_64 0.14391156830900597f   // ln(10000)/64
#define SM_SCALE        0.08838834764831845f   // 1/sqrt(128)

__global__ __launch_bounds__(256) void rope_pack_q(const float* __restrict__ Qraw,
                                                   _Float16* __restrict__ Qb) {
  int idx = blockIdx.x * 256 + threadIdx.x;          // S*32*64 threads
  if (idx >= S_LEN * NQH * 64) return;
  int i = idx & 63;
  int h = (idx >> 6) & 31;
  int s = idx >> 11;
  const float* src = Qraw + (size_t)s * HID + h * DHEAD;
  float q0 = src[i], q1 = src[i + 64];
  float inv = __expf(-(float)i * LN10000_OVER_64);
  float sn, cs;
  __sincosf((float)s * inv, &sn, &cs);
  _Float16* dst = Qb + ((size_t)h * S_LEN + s) * DHEAD;
  dst[i]      = (_Float16)((q0 * cs - q1 * sn) * SM_SCALE);
  dst[i + 64] = (_Float16)((q1 * cs + q0 * sn) * SM_SCALE);
}

__global__ __launch_bounds__(256) void rope_pack_k(const float* __restrict__ Kraw,
                                                   _Float16* __restrict__ Kb) {
  int idx = blockIdx.x * 256 + threadIdx.x;          // S*8*64 threads
  if (idx >= S_LEN * NKVH * 64) return;
  int i = idx & 63;
  int h = (idx >> 6) & 7;
  int s = idx >> 9;
  const float* src = Kraw + (size_t)s * (NKVH * DHEAD) + h * DHEAD;
  float k0 = src[i], k1 = src[i + 64];
  float inv = __expf(-(float)i * LN10000_OVER_64);
  float sn, cs;
  __sincosf((float)s * inv, &sn, &cs);
  _Float16* dst = Kb + ((size_t)h * S_LEN + s) * DHEAD;
  dst[i]      = (_Float16)(k0 * cs - k1 * sn);
  dst[i + 64] = (_Float16)(k1 * cs + k0 * sn);
}

// V packed d-major: Vb[kv][d][s] so PV B-fragments read contiguous rows.
__global__ __launch_bounds__(256) void pack_v(const float* __restrict__ Vraw,
                                              _Float16* __restrict__ Vb) {
  int idx = blockIdx.x * 256 + threadIdx.x;          // S*8*128 threads
  if (idx >= S_LEN * NKVH * DHEAD) return;
  int s = idx & (S_LEN - 1);
  int d = (idx >> 11) & 127;
  int h = idx >> 18;
  Vb[((size_t)h * DHEAD + d) * S_LEN + s] =
      (_Float16)Vraw[(size_t)s * (NKVH * DHEAD) + h * DHEAD + d];
}

// ---------------------------------------------------------------------------
// Fused flash attention (causal, GQA 4:1).
// Grid: (S/64 q-blocks, 32 heads). Block: 128 threads = 4 waves.
// Wave w owns q rows [qBase + 16w, qBase + 16w + 16).
// Per 32-key chunk: 8 WMMA (QK^T) + online softmax + 8 WMMA (PV).
// ---------------------------------------------------------------------------
__global__ __launch_bounds__(128) void flash_attn(const _Float16* __restrict__ Qb,
                                                  const _Float16* __restrict__ Kb,
                                                  const _Float16* __restrict__ Vb,
                                                  _Float16* __restrict__ Ob) {
  __shared__ __align__(16) _Float16 Kt[32][136];     // 32 keys x 128 d (padded)
  __shared__ __align__(16) _Float16 Vt[128][40];     // 128 d x 32 keys (padded)
  __shared__ __align__(16) _Float16 Ps[4][16][40];   // per-wave P transpose

  const int qh    = blockIdx.y;
  const int kv    = qh >> 2;                         // GQA group
  const int qBase = blockIdx.x * 64;
  const int tid   = threadIdx.x;
  const int lane  = tid & 31;
  const int w     = tid >> 5;
  const int ln15  = lane & 15;
  const int hi    = lane >> 4;
  const int kBa   = hi ? 8 : 0;
  const int kBb   = hi ? 16 : 0;

  // preload Q A-fragments (16x128 per wave -> 4 frags of K=32)
  const int qRowA = qBase + w * 16 + ln15;
  const _Float16* qrow = Qb + ((size_t)qh * S_LEN + qRowA) * DHEAD;
  v16h qf[4];
#pragma unroll
  for (int dc = 0; dc < 4; ++dc) qf[dc] = load_frag_a(qrow + dc * 32, kBa);

  v8f accO[8];
#pragma unroll
  for (int f = 0; f < 8; ++f) { v8f z = {}; accO[f] = z; }
  float mst[8], lst[8];
#pragma unroll
  for (int r = 0; r < 8; ++r) { mst[r] = -INFINITY; lst[r] = 0.0f; }

  const int kChunks = (qBase + 64) >> 5;             // causal: keys < qBase+64
  for (int kc = 0; kc < kChunks; ++kc) {
    const int kBase = kc * 32;
    __syncthreads();
#if HAVE_ASYNC_LDS
    // K tile (32 keys x 128 d) + V tile transposed (128 d x 32 keys)
#pragma unroll
    for (int c = 0; c < 4; ++c) {
      int ci = tid + c * 128;
      int kk = ci >> 4;
      int c8 = (ci & 15) * 8;
      async_cp16(&Kt[kk][c8],
                 Kb + ((size_t)kv * S_LEN + kBase + kk) * DHEAD + c8);
      int d  = ci >> 2;
      int c4 = (ci & 3) * 8;
      async_cp16(&Vt[d][c4],
                 Vb + ((size_t)kv * DHEAD + d) * S_LEN + kBase + c4);
    }
    wait_async0();
#else
    h8 tk[4], tv[4];
#pragma unroll
    for (int c = 0; c < 4; ++c) {
      int ci = tid + c * 128;
      int kk = ci >> 4;
      int c8 = (ci & 15) * 8;
      tk[c] = *(const h8*)(Kb + ((size_t)kv * S_LEN + kBase + kk) * DHEAD + c8);
      int d  = ci >> 2;
      int c4 = (ci & 3) * 8;
      tv[c] = *(const h8*)(Vb + ((size_t)kv * DHEAD + d) * S_LEN + kBase + c4);
    }
#pragma unroll
    for (int c = 0; c < 4; ++c) {
      int ci = tid + c * 128;
      int kk = ci >> 4;
      int c8 = (ci & 15) * 8;
      *(h8*)&Kt[kk][c8] = tk[c];
      int d  = ci >> 2;
      int c4 = (ci & 3) * 8;
      *(h8*)&Vt[d][c4] = tv[c];
    }
#endif
    __syncthreads();

    if (qBase + w * 16 + 15 >= kBase) {              // wave-uniform skip
      // S = Q K^T : two 16-key column tiles, K-dim = 128 (4 chunks of 32)
      v8f sc[2];
#pragma unroll
      for (int t = 0; t < 2; ++t) {
        v8f z = {}; sc[t] = z;
#pragma unroll
        for (int dc = 0; dc < 4; ++dc) {
          v16h kf = load_frag_b(&Kt[t * 16 + ln15][dc * 32 + kBb]);
          sc[t] = wmma_f16(qf[dc], kf, sc[t]);
        }
      }

      // online softmax over this 32-key chunk (row stats per C-frag VGPR)
#pragma unroll
      for (int r = 0; r < 8; ++r) {
        int qRow = qBase + w * 16 + r + hi * 8;
        float s0 = sc[0][r];
        float s1 = sc[1][r];
        if (kBase + ln15 > qRow)      s0 = -INFINITY;   // causal mask
        if (kBase + 16 + ln15 > qRow) s1 = -INFINITY;
        float v = fmaxf(s0, s1);
#pragma unroll
        for (int off = 1; off < 16; off <<= 1)
          v = fmaxf(v, __shfl_xor(v, off, 32));         // within 16-lane half
        float mn    = fmaxf(mst[r], v);
        float alpha = __expf(mst[r] - mn);
        float p0 = __expf(s0 - mn);
        float p1 = __expf(s1 - mn);
        float rs = p0 + p1;
#pragma unroll
        for (int off = 1; off < 16; off <<= 1)
          rs += __shfl_xor(rs, off, 32);
        lst[r] = lst[r] * alpha + rs;
        mst[r] = mn;
#pragma unroll
        for (int f = 0; f < 8; ++f)
          accO[f][r] = accO[f][r] * alpha;
        // C-layout -> A-layout transpose through per-wave LDS (in-order DS)
        Ps[w][r + hi * 8][ln15]      = (_Float16)p0;
        Ps[w][r + hi * 8][16 + ln15] = (_Float16)p1;
      }

      // O += P (16x32) x V (32x128)
      v16h pf = load_frag_a(&Ps[w][ln15][0], kBa);
#pragma unroll
      for (int f = 0; f < 8; ++f) {
        v16h vf = load_frag_b(&Vt[f * 16 + ln15][kBb]);
        accO[f] = wmma_f16(pf, vf, accO[f]);
      }
    }
  }

  // normalize + write half output [S, 4096]
#pragma unroll
  for (int r = 0; r < 8; ++r) {
    float inv = 1.0f / lst[r];
    int qRow = qBase + w * 16 + r + hi * 8;
    _Float16* orow = Ob + (size_t)qRow * HID + qh * DHEAD;
#pragma unroll
    for (int f = 0; f < 8; ++f)
      orow[f * 16 + ln15] = (_Float16)(accO[f][r] * inv);
  }
}

// ---------------------------------------------------------------------------
// Host launcher
// ---------------------------------------------------------------------------
extern "C" void kernel_launch(void* const* d_in, const int* in_sizes, int n_in,
                              void* d_out, int out_size, void* d_ws, size_t ws_size,
                              hipStream_t stream) {
  (void)in_sizes; (void)n_in; (void)out_size; (void)ws_size;
  const float* hs = (const float*)d_in[0];
  const float* Wq = (const float*)d_in[1];
  const float* Wk = (const float*)d_in[2];
  const float* Wv = (const float*)d_in[3];
  const float* Wo = (const float*)d_in[4];
  float* out = (float*)d_out;

  char* ws = (char*)d_ws;
  size_t off = 0;
  _Float16* Xb  = (_Float16*)(ws + off); off += (size_t)S_LEN * HID * 2;                 // 16 MiB
  _Float16* Wqb = (_Float16*)(ws + off); off += (size_t)HID * HID * 2;                   // 32 MiB
  _Float16* Wkb = (_Float16*)(ws + off); off += (size_t)(NKVH * DHEAD) * HID * 2;        //  8 MiB
  _Float16* Wvb = (_Float16*)(ws + off); off += (size_t)(NKVH * DHEAD) * HID * 2;        //  8 MiB
  _Float16* Wob = (_Float16*)(ws + off); off += (size_t)HID * HID * 2;                   // 32 MiB
  float*    Qraw = (float*)(ws + off);   off += (size_t)S_LEN * HID * 4;                 // 32 MiB
  float*    Kraw = (float*)(ws + off);   off += (size_t)S_LEN * (NKVH * DHEAD) * 4;      //  8 MiB
  float*    Vraw = (float*)(ws + off);   off += (size_t)S_LEN * (NKVH * DHEAD) * 4;      //  8 MiB
  _Float16* Qb  = (_Float16*)(ws + off); off += (size_t)NQH * S_LEN * DHEAD * 2;         // 16 MiB
  _Float16* Kb  = (_Float16*)(ws + off); off += (size_t)NKVH * S_LEN * DHEAD * 2;        //  4 MiB
  _Float16* Vb  = (_Float16*)(ws + off); off += (size_t)NKVH * DHEAD * S_LEN * 2;        //  4 MiB
  _Float16* Ob  = (_Float16*)(ws + off); off += (size_t)S_LEN * HID * 2;                 // 16 MiB

  // 1) convert inputs to f16
  {
    int nX  = S_LEN * HID;
    int nWq = HID * HID;
    int nWk = (NKVH * DHEAD) * HID;
    cvt_f32_to_f16<<<(nX  + 255) / 256, 256, 0, stream>>>(hs, Xb,  nX);
    cvt_f32_to_f16<<<(nWq + 255) / 256, 256, 0, stream>>>(Wq, Wqb, nWq);
    cvt_f32_to_f16<<<(nWk + 255) / 256, 256, 0, stream>>>(Wk, Wkb, nWk);
    cvt_f32_to_f16<<<(nWk + 255) / 256, 256, 0, stream>>>(Wv, Wvb, nWk);
    cvt_f32_to_f16<<<(nWq + 255) / 256, 256, 0, stream>>>(Wo, Wob, nWq);
  }

  // 2) QKV projections (y = x W^T, NT GEMM), block tile 128x128
  gemm_nt<<<dim3(HID / 128, S_LEN / 128), 256, 0, stream>>>(Xb, Wqb, Qraw, S_LEN, HID, HID);
  gemm_nt<<<dim3((NKVH * DHEAD) / 128, S_LEN / 128), 256, 0, stream>>>(Xb, Wkb, Kraw, S_LEN, NKVH * DHEAD, HID);
  gemm_nt<<<dim3((NKVH * DHEAD) / 128, S_LEN / 128), 256, 0, stream>>>(Xb, Wvb, Vraw, S_LEN, NKVH * DHEAD, HID);

  // 3) RoPE + pack to head-major f16 (Q carries softmax scale)
  rope_pack_q<<<(S_LEN * NQH * 64) / 256, 256, 0, stream>>>(Qraw, Qb);
  rope_pack_k<<<(S_LEN * NKVH * 64) / 256, 256, 0, stream>>>(Kraw, Kb);
  pack_v<<<(S_LEN * NKVH * DHEAD) / 256, 256, 0, stream>>>(Vraw, Vb);

  // 4) fused causal flash attention
  flash_attn<<<dim3(S_LEN / 64, NQH), 128, 0, stream>>>(Qb, Kb, Vb, Ob);

  // 5) output projection -> fp32 d_out
  gemm_nt<<<dim3(HID / 128, S_LEN / 128), 256, 0, stream>>>(Ob, Wob, out, S_LEN, HID, HID);
}
